// Model_65163243815576
// MI455X (gfx1250) — matrix operand order repaired and checked by reading, not compiled, not run on hardware
//
#include <hip/hip_runtime.h>

#define N_NODES 15000
#define N_EDGES 90000
#define LAT 128
#define NBLK 15
#define STR 136   // staging stride: 136*2=272B (16B aligned), 68 dwords % 64 = 4 -> bank stagger

typedef __attribute__((ext_vector_type(16))) _Float16 v16h;
typedef __attribute__((ext_vector_type(8)))  float    v8f;
typedef __attribute__((ext_vector_type(4)))  float    f4;

// ---------------------------------------------------------------------------
// WMMA helpers (wave32; v_wmma_f32_16x16x32_f16)
// ---------------------------------------------------------------------------

// Cooperative copy of a transposed f16 weight matrix [128][K] (global) into
// LDS with padded row stride KP (halfs). 16B chunks, 256 threads.
template<int K, int KP>
__device__ __forceinline__ void load_weights_lds(_Float16* sW, const _Float16* __restrict__ Wt)
{
  constexpr int CH = K / 8;                 // 16B chunks per row
  const int t = threadIdx.x;
#pragma unroll 4
  for (int c = t; c < 128 * CH; c += 256) {
    const int n  = c / CH;
    const int k8 = (c % CH) * 8;
    *(f4*)(sW + n * KP + k8) = *(const f4*)(Wt + (size_t)n * K + k8);
  }
}

// One 128-wide FC layer for a 16-row tile:
//   D[16x128] = A[16x(KB*32)] x W[(KB*32)x128] + bias, optional ReLU.
// B comes from LDS (sW, padded stride KP). All 8 N-tile accumulators are kept
// live; per k-step we issue 16 ds_load_b128 then 8 independent WMMAs so the
// scheduler can pipeline loads against the matrix pipe (no full-drain waits).
// getA(kb) supplies the A tile in WMMA A layout:
//   lanes 0-15: row M=lane, K {0..7}+{16..23}; lanes 16-31: row M=lane-16, K {8..15}+{24..31}.
// Result written row-major f16 into per-wave LDS staging (stride STR).
template<int KB, int KP, typename GetA>
__device__ __forceinline__ void wmma_fc(GetA getA, const _Float16* sW,
                                        const float* __restrict__ bias,
                                        _Float16* sOut, bool relu)
{
  const int lane = threadIdx.x & 31;
  const int nlo = lane & 15, hi = lane >> 4;
  v8f acc[8];
#pragma unroll
  for (int nt = 0; nt < 8; ++nt) {
    const float bv = bias[nt * 16 + nlo];
    acc[nt] = (v8f){bv, bv, bv, bv, bv, bv, bv, bv};
  }
#pragma unroll
  for (int kb = 0; kb < KB; ++kb) {
    const v16h a = getA(kb);
    v16h b[8];
#pragma unroll
    for (int nt = 0; nt < 8; ++nt) {
      // B layout: lanes 0-15 col N, K {0..7}+{16..23}; lanes 16-31 col N, K {8..15}+{24..31}.
      const _Float16* p = sW + (nt * 16 + nlo) * KP + kb * 32 + hi * 8;
      ((f4*)&b[nt])[0] = *(const f4*)(p);
      ((f4*)&b[nt])[1] = *(const f4*)(p + 16);
    }
#pragma unroll
    for (int nt = 0; nt < 8; ++nt)
      acc[nt] = __builtin_amdgcn_wmma_f32_16x16x32_f16(
          false, a, false, b[nt], (short)0, acc[nt], false, false);
  }
#pragma unroll
  for (int nt = 0; nt < 8; ++nt) {
#pragma unroll
    for (int r = 0; r < 8; ++r) {
      float v = acc[nt][r];
      if (relu) v = fmaxf(v, 0.0f);
      // C/D layout: lane col n; VGPR r -> row r + (lane>=16 ? 8 : 0)
      sOut[(r + hi * 8) * STR + nt * 16 + nlo] = (_Float16)v;
    }
  }
}

// LayerNorm over 128 cols of a 16-row staged tile; 2 lanes per row (64 cols each).
__device__ __forceinline__ void layernorm_tile(_Float16* s, const float* __restrict__ g,
                                               const float* __restrict__ b)
{
  const int lane = threadIdx.x & 31;
  const int row = lane & 15, half = lane >> 4;
  _Float16* p = s + row * STR + half * 64;
  float sum = 0.f, sq = 0.f;
#pragma unroll 8
  for (int c = 0; c < 64; ++c) { float v = (float)p[c]; sum += v; sq += v * v; }
  sum += __shfl_xor(sum, 16, 32);
  sq  += __shfl_xor(sq, 16, 32);
  const float mu  = sum * (1.f / 128.f);
  const float var = sq * (1.f / 128.f) - mu * mu;
  const float rs  = rsqrtf(var + 1e-5f);
#pragma unroll 8
  for (int c = 0; c < 64; ++c) {
    const int col = half * 64 + c;
    p[c] = (_Float16)(((float)p[c] - mu) * rs * g[col] + b[col]);
  }
}

// ---------------------------------------------------------------------------
// Small scalar kernels
// ---------------------------------------------------------------------------

__global__ void transpose_cvt_kernel(const float* __restrict__ in, _Float16* __restrict__ out, int K)
{ // in: f32 [K][128]  ->  out: f16 [128][K]
  int id = blockIdx.x * blockDim.x + threadIdx.x;
  if (id >= K * 128) return;
  int n = id / K, k = id % K;
  out[id] = (_Float16)in[(size_t)k * 128 + n];
}

__global__ void zero_kernel(float* p, int n) {
  int i = blockIdx.x * blockDim.x + threadIdx.x;
  if (i < n) p[i] = 0.f;
}

__global__ void node_features_kernel(const float* __restrict__ wp, const float* __restrict__ pwp,
                                     const int* __restrict__ ntype,
                                     const float* __restrict__ mean, const float* __restrict__ std,
                                     float* __restrict__ out, int n)
{
  int i = blockIdx.x * blockDim.x + threadIdx.x;
  if (i >= n) return;
  float f[12];
  f[0] = wp[i * 3 + 0] - pwp[i * 3 + 0];
  f[1] = wp[i * 3 + 1] - pwp[i * 3 + 1];
  f[2] = wp[i * 3 + 2] - pwp[i * 3 + 2];
#pragma unroll
  for (int t = 0; t < 9; ++t) f[3 + t] = 0.f;
  int t = ntype[i];
  if (t >= 0 && t < 9) f[3 + t] = 1.f;
#pragma unroll
  for (int j = 0; j < 12; ++j) out[(size_t)i * 12 + j] = (f[j] - mean[j]) / std[j];
}

__global__ void edge_features_kernel(const float* __restrict__ wp, const float* __restrict__ mp,
                                     const int* __restrict__ snd, const int* __restrict__ rcv,
                                     const float* __restrict__ mean, const float* __restrict__ std,
                                     float* __restrict__ out, int n)
{
  int i = blockIdx.x * blockDim.x + threadIdx.x;
  if (i >= n) return;
  int s = snd[i], r = rcv[i];
  float f[7];
  f[0] = wp[s * 3 + 0] - wp[r * 3 + 0];
  f[1] = wp[s * 3 + 1] - wp[r * 3 + 1];
  f[2] = wp[s * 3 + 2] - wp[r * 3 + 2];
  f[3] = sqrtf(f[0] * f[0] + f[1] * f[1] + f[2] * f[2]);
  f[4] = mp[s * 2 + 0] - mp[r * 2 + 0];
  f[5] = mp[s * 2 + 1] - mp[r * 2 + 1];
  f[6] = sqrtf(f[4] * f[4] + f[5] * f[5]);
#pragma unroll
  for (int j = 0; j < 7; ++j) out[(size_t)i * 7 + j] = (f[j] - mean[j]) / std[j];
}

// First (skinny) encoder layer: [rows][K] f32 @ W[K][128] + b, ReLU -> f16
__global__ void fc_small_kernel(const float* __restrict__ in, const float* __restrict__ W,
                                const float* __restrict__ b, _Float16* __restrict__ out,
                                int rows, int K)
{
  int id = blockIdx.x * blockDim.x + threadIdx.x;
  if (id >= rows * 128) return;
  int row = id >> 7, c = id & 127;
  const float* x = in + (size_t)row * K;
  float acc = b[c];
  for (int k = 0; k < K; ++k) acc += x[k] * W[k * 128 + c];
  out[id] = (_Float16)fmaxf(acc, 0.f);
}

// Decoder last layer (128->3) + out-normalizer + 2nd-order integration + mask
__global__ void decode_integrate_kernel(const _Float16* __restrict__ h,
                                        const float* __restrict__ W, const float* __restrict__ b,
                                        const float* __restrict__ om_mean, const float* __restrict__ om_std,
                                        const float* __restrict__ wp, const float* __restrict__ pwp,
                                        const int* __restrict__ ntype, float* __restrict__ out, int n)
{
  int i = blockIdx.x * blockDim.x + threadIdx.x;
  if (i >= n) return;
  const _Float16* x = h + (size_t)i * 128;
  float p0 = b[0], p1 = b[1], p2 = b[2];
  for (int k = 0; k < 128; ++k) {
    float v = (float)x[k];
    p0 += v * W[k * 3 + 0];
    p1 += v * W[k * 3 + 1];
    p2 += v * W[k * 3 + 2];
  }
  float a0 = p0 * om_std[0] + om_mean[0];
  float a1 = p1 * om_std[1] + om_mean[1];
  float a2 = p2 * om_std[2] + om_mean[2];
  bool nm = (ntype[i] == 0);
  out[i * 3 + 0] = nm ? (2.f * wp[i * 3 + 0] + a0 - pwp[i * 3 + 0]) : wp[i * 3 + 0];
  out[i * 3 + 1] = nm ? (2.f * wp[i * 3 + 1] + a1 - pwp[i * 3 + 1]) : wp[i * 3 + 1];
  out[i * 3 + 2] = nm ? (2.f * wp[i * 3 + 2] + a2 - pwp[i * 3 + 2]) : wp[i * 3 + 2];
}

// ---------------------------------------------------------------------------
// Fused WMMA MLP kernels (LDS-resident weights, per-wave staging)
// ---------------------------------------------------------------------------

// 2-layer 128->128->128 WMMA MLP (layer1 ReLU; layer2 ReLU if RELU2; LN if LN).
template<bool RELU2, bool LN>
__global__ __launch_bounds__(256) void mlp2_wmma_kernel(
    const _Float16* __restrict__ in, const _Float16* __restrict__ Wt2, const float* __restrict__ b2,
    const _Float16* __restrict__ Wt3, const float* __restrict__ b3,
    const float* __restrict__ g, const float* __restrict__ bb,
    _Float16* __restrict__ out, int rows)
{
  __shared__ __align__(16) _Float16 sW[128 * 136];
  __shared__ __align__(16) _Float16 stage[8][16][STR];
  const int wave = threadIdx.x >> 5, lane = threadIdx.x & 31;
  const int tile = blockIdx.x * 8 + wave;
  const bool active = (tile * 16 < rows);
  const int r0 = active ? tile * 16 : 0;
  const int m = lane & 15, hi = lane >> 4;
  int row = r0 + m; if (row >= rows) row = rows - 1;
  _Float16* st = &stage[wave][0][0];

  const _Float16* inrow = in + (size_t)row * 128;
  auto getGlobalA = [&](int kb) -> v16h {
    v16h a;
    const _Float16* p = inrow + kb * 32 + hi * 8;
    ((f4*)&a)[0] = *(const f4*)(p);
    ((f4*)&a)[1] = *(const f4*)(p + 16);
    return a;
  };
  auto getStageA = [&](int kb) -> v16h {
    v16h a;
    const _Float16* p = st + m * STR + kb * 32 + hi * 8;
    ((f4*)&a)[0] = *(const f4*)(p);
    ((f4*)&a)[1] = *(const f4*)(p + 16);
    return a;
  };

  load_weights_lds<128, 136>(sW, Wt2);
  __syncthreads();
  wmma_fc<4, 136>(getGlobalA, sW, b2, st, true);
  __syncthreads();
  load_weights_lds<128, 136>(sW, Wt3);
  __syncthreads();
  wmma_fc<4, 136>(getStageA, sW, b3, st, RELU2);
  if (LN) layernorm_tile(st, g, bb);

  const int orow = r0 + m;
  if (active && orow < rows) {
    const _Float16* sp = st + m * STR + hi * 64;
    _Float16* op = out + (size_t)orow * 128 + hi * 64;
#pragma unroll 16
    for (int c = 0; c < 64; ++c) op[c] = sp[c];
  }
}

// Edge block: gather-concat [edge_lat | nlat[s] | nlat[r]] (K=384) -> 3-layer MLP
// + LN, residual update of edge_lat, fused segment_sum into agg (f32 atomics).
__global__ __launch_bounds__(256) void edge_block_kernel(
    const _Float16* __restrict__ edge_lat, const _Float16* __restrict__ node_lat,
    const int* __restrict__ snd, const int* __restrict__ rcv,
    const _Float16* __restrict__ Wt1, const float* __restrict__ b1,
    const _Float16* __restrict__ Wt2, const float* __restrict__ b2,
    const _Float16* __restrict__ Wt3, const float* __restrict__ b3,
    const float* __restrict__ g, const float* __restrict__ bb,
    _Float16* __restrict__ edge_out, float* __restrict__ agg, int nE)
{
  __shared__ __align__(16) _Float16 sW[128 * 392];   // fits K=384 (KP=392), reused for K=128
  __shared__ __align__(16) _Float16 stage[8][16][STR];
  const int wave = threadIdx.x >> 5, lane = threadIdx.x & 31;
  const int tile = blockIdx.x * 8 + wave;
  const bool active = (tile * 16 < nE);
  const int e0 = active ? tile * 16 : 0;
  const int m = lane & 15, hi = lane >> 4;
  int e = e0 + m; if (e >= nE) e = nE - 1;
  const int s = snd[e], r = rcv[e];
  _Float16* st = &stage[wave][0][0];

  const _Float16* src[3] = { edge_lat + (size_t)e * 128,
                             node_lat + (size_t)s * 128,
                             node_lat + (size_t)r * 128 };
  auto getGatherA = [&](int kb) -> v16h {
    v16h a;
    const _Float16* p = src[kb >> 2] + (kb & 3) * 32 + hi * 8;
    ((f4*)&a)[0] = *(const f4*)(p);
    ((f4*)&a)[1] = *(const f4*)(p + 16);
    return a;
  };
  auto getStageA = [&](int kb) -> v16h {
    v16h a;
    const _Float16* p = st + m * STR + kb * 32 + hi * 8;
    ((f4*)&a)[0] = *(const f4*)(p);
    ((f4*)&a)[1] = *(const f4*)(p + 16);
    return a;
  };

  load_weights_lds<384, 392>(sW, Wt1);
  __syncthreads();
  wmma_fc<12, 392>(getGatherA, sW, b1, st, true);
  __syncthreads();
  load_weights_lds<128, 136>(sW, Wt2);
  __syncthreads();
  wmma_fc<4, 136>(getStageA, sW, b2, st, true);
  __syncthreads();
  load_weights_lds<128, 136>(sW, Wt3);
  __syncthreads();
  wmma_fc<4, 136>(getStageA, sW, b3, st, false);
  layernorm_tile(st, g, bb);

  // Residual + fused segment-sum (2 lanes per edge, 64 cols each).
  const int e2 = e0 + m;
  if (active && e2 < nE) {
    const int dest = rcv[e2];
    const _Float16* sp = st + m * STR + hi * 64;
    const _Float16* ip = edge_lat + (size_t)e2 * 128 + hi * 64;
    _Float16* op = edge_out + (size_t)e2 * 128 + hi * 64;
    float* ap = agg + (size_t)dest * 128 + hi * 64;
#pragma unroll 8
    for (int c = 0; c < 64; ++c) {
      float nv = (float)sp[c];
      atomicAdd(&ap[c], nv);
      op[c] = (_Float16)((float)ip[c] + nv);
    }
  }
}

// Node block: concat [node_lat | agg] (K=256) -> 3-layer MLP + LN + residual.
__global__ __launch_bounds__(256) void node_block_kernel(
    const _Float16* __restrict__ node_lat, const float* __restrict__ agg,
    const _Float16* __restrict__ Wt1, const float* __restrict__ b1,
    const _Float16* __restrict__ Wt2, const float* __restrict__ b2,
    const _Float16* __restrict__ Wt3, const float* __restrict__ b3,
    const float* __restrict__ g, const float* __restrict__ bb,
    _Float16* __restrict__ node_out, int nN)
{
  __shared__ __align__(16) _Float16 sW[128 * 264];   // fits K=256 (KP=264), reused for K=128
  __shared__ __align__(16) _Float16 stage[8][16][STR];
  const int wave = threadIdx.x >> 5, lane = threadIdx.x & 31;
  const int tile = blockIdx.x * 8 + wave;
  const bool active = (tile * 16 < nN);
  const int r0 = active ? tile * 16 : 0;
  const int m = lane & 15, hi = lane >> 4;
  int row = r0 + m; if (row >= nN) row = nN - 1;
  _Float16* st = &stage[wave][0][0];

  const _Float16* nrow = node_lat + (size_t)row * 128;
  const float*    arow = agg + (size_t)row * 128;
  auto getConcatA = [&](int kb) -> v16h {
    v16h a;
    if (kb < 4) {
      const _Float16* p = nrow + kb * 32 + hi * 8;
      ((f4*)&a)[0] = *(const f4*)(p);
      ((f4*)&a)[1] = *(const f4*)(p + 16);
    } else {
      const int k0 = (kb - 4) * 32 + hi * 8;
      float t[16];
      *(f4*)&t[0]  = *(const f4*)(arow + k0);
      *(f4*)&t[4]  = *(const f4*)(arow + k0 + 4);
      *(f4*)&t[8]  = *(const f4*)(arow + k0 + 16);
      *(f4*)&t[12] = *(const f4*)(arow + k0 + 20);
#pragma unroll
      for (int i = 0; i < 16; ++i) a[i] = (_Float16)t[i];
    }
    return a;
  };
  auto getStageA = [&](int kb) -> v16h {
    v16h a;
    const _Float16* p = st + m * STR + kb * 32 + hi * 8;
    ((f4*)&a)[0] = *(const f4*)(p);
    ((f4*)&a)[1] = *(const f4*)(p + 16);
    return a;
  };

  load_weights_lds<256, 264>(sW, Wt1);
  __syncthreads();
  wmma_fc<8, 264>(getConcatA, sW, b1, st, true);
  __syncthreads();
  load_weights_lds<128, 136>(sW, Wt2);
  __syncthreads();
  wmma_fc<4, 136>(getStageA, sW, b2, st, true);
  __syncthreads();
  load_weights_lds<128, 136>(sW, Wt3);
  __syncthreads();
  wmma_fc<4, 136>(getStageA, sW, b3, st, false);
  layernorm_tile(st, g, bb);

  const int orow = r0 + m;
  if (active && orow < nN) {
    const _Float16* sp = st + m * STR + hi * 64;
    const _Float16* ip = node_lat + (size_t)orow * 128 + hi * 64;
    _Float16* op = node_out + (size_t)orow * 128 + hi * 64;
#pragma unroll 16
    for (int c = 0; c < 64; ++c) op[c] = (_Float16)((float)ip[c] + (float)sp[c]);
  }
}

// ---------------------------------------------------------------------------
// Host launch
// ---------------------------------------------------------------------------
// Input flattening (jax pytree: dict keys sorted, lists in order):
//   0..5: world_pos, prev_world_pos, mesh_pos, node_type, senders, receivers
//   params: blocks[15]{edge,node}, dec, edge_norm, enc_edge, enc_node, node_norm, out_norm
//   MLP leaves (ln=True): [W0,W1,W2,b0,b1,b2,ln_b,ln_g]   ("Ws"<"bs"<"ln_b"<"ln_g")
//   dec (ln=False):       [W0,W1,W2,b0,b1,b2]
//   normalizer:           [mean,std]

extern "C" void kernel_launch(void* const* d_in, const int* in_sizes, int n_in,
                              void* d_out, int out_size, void* d_ws, size_t ws_size,
                              hipStream_t stream)
{
  (void)in_sizes; (void)n_in; (void)out_size; (void)ws_size;
  const int N = N_NODES, E = N_EDGES;

  const float* wp   = (const float*)d_in[0];
  const float* pwp  = (const float*)d_in[1];
  const float* mp   = (const float*)d_in[2];
  const int*  ntype = (const int*)d_in[3];
  const int*  snd   = (const int*)d_in[4];
  const int*  rcv   = (const int*)d_in[5];

  const int P = 6;
  const int D = P + NBLK * 16;               // dec base
  const float* decW0 = (const float*)d_in[D + 0];
  const float* decW1 = (const float*)d_in[D + 1];
  const float* decW2 = (const float*)d_in[D + 2];
  const float* decb0 = (const float*)d_in[D + 3];
  const float* decb1 = (const float*)d_in[D + 4];
  const float* decb2 = (const float*)d_in[D + 5];
  const float* edgeN_mean = (const float*)d_in[D + 6];
  const float* edgeN_std  = (const float*)d_in[D + 7];
  const int EE = D + 8;                       // enc_edge
  const int EN = D + 16;                      // enc_node
  const float* nodeN_mean = (const float*)d_in[D + 24];
  const float* nodeN_std  = (const float*)d_in[D + 25];
  const float* outN_mean  = (const float*)d_in[D + 26];
  const float* outN_std   = (const float*)d_in[D + 27];

  // ---- workspace carve ----
  char* base = (char*)d_ws;
  size_t off = 0;
  auto alloc = [&](size_t bytes) -> void* {
    void* p = base + off;
    off += (bytes + 255) & ~(size_t)255;
    return p;
  };
  _Float16* encE_W2t = (_Float16*)alloc(128 * 128 * 2);
  _Float16* encE_W3t = (_Float16*)alloc(128 * 128 * 2);
  _Float16* encN_W2t = (_Float16*)alloc(128 * 128 * 2);
  _Float16* encN_W3t = (_Float16*)alloc(128 * 128 * 2);
  _Float16* dec_W1t  = (_Float16*)alloc(128 * 128 * 2);
  _Float16* dec_W2t  = (_Float16*)alloc(128 * 128 * 2);
  _Float16 *eW1t[NBLK], *eW2t[NBLK], *eW3t[NBLK], *nW1t[NBLK], *nW2t[NBLK], *nW3t[NBLK];
  for (int i = 0; i < NBLK; ++i) {
    eW1t[i] = (_Float16*)alloc(384 * 128 * 2);
    eW2t[i] = (_Float16*)alloc(128 * 128 * 2);
    eW3t[i] = (_Float16*)alloc(128 * 128 * 2);
    nW1t[i] = (_Float16*)alloc(256 * 128 * 2);
    nW2t[i] = (_Float16*)alloc(128 * 128 * 2);
    nW3t[i] = (_Float16*)alloc(128 * 128 * 2);
  }
  float*    node_feat = (float*)alloc((size_t)N * 12 * 4);
  float*    edge_feat = (float*)alloc((size_t)E * 7 * 4);
  _Float16* h1        = (_Float16*)alloc((size_t)E * 128 * 2);   // reused buffer
  _Float16* node_lat  = (_Float16*)alloc((size_t)N * 128 * 2);
  _Float16* edge_lat  = (_Float16*)alloc((size_t)E * 128 * 2);
  float*    agg       = (float*)alloc((size_t)N * 128 * 4);

  // ---- weight convert+transpose (f32 [K][128] -> f16 [128][K]) ----
  auto tcvt = [&](const float* in, _Float16* out, int K) {
    int total = K * 128;
    transpose_cvt_kernel<<<(total + 255) / 256, 256, 0, stream>>>(in, out, K);
  };
  tcvt((const float*)d_in[EE + 1], encE_W2t, 128);
  tcvt((const float*)d_in[EE + 2], encE_W3t, 128);
  tcvt((const float*)d_in[EN + 1], encN_W2t, 128);
  tcvt((const float*)d_in[EN + 2], encN_W3t, 128);
  tcvt(decW0, dec_W1t, 128);
  tcvt(decW1, dec_W2t, 128);
  for (int i = 0; i < NBLK; ++i) {
    const int BE = P + i * 16, BN = P + i * 16 + 8;
    tcvt((const float*)d_in[BE + 0], eW1t[i], 384);
    tcvt((const float*)d_in[BE + 1], eW2t[i], 128);
    tcvt((const float*)d_in[BE + 2], eW3t[i], 128);
    tcvt((const float*)d_in[BN + 0], nW1t[i], 256);
    tcvt((const float*)d_in[BN + 1], nW2t[i], 128);
    tcvt((const float*)d_in[BN + 2], nW3t[i], 128);
  }

  const int nodeBlocks = (((N + 15) / 16) + 7) / 8;  // 16-row waves, 8 waves/block
  const int edgeBlocks = (((E + 15) / 16) + 7) / 8;

  // ---- encode nodes ----
  node_features_kernel<<<(N + 255) / 256, 256, 0, stream>>>(
      wp, pwp, ntype, nodeN_mean, nodeN_std, node_feat, N);
  fc_small_kernel<<<(N * 128 + 255) / 256, 256, 0, stream>>>(
      node_feat, (const float*)d_in[EN + 0], (const float*)d_in[EN + 3], h1, N, 12);
  mlp2_wmma_kernel<false, true><<<nodeBlocks, 256, 0, stream>>>(
      h1, encN_W2t, (const float*)d_in[EN + 4], encN_W3t, (const float*)d_in[EN + 5],
      (const float*)d_in[EN + 7], (const float*)d_in[EN + 6], node_lat, N);

  // ---- encode edges ----
  edge_features_kernel<<<(E + 255) / 256, 256, 0, stream>>>(
      wp, mp, snd, rcv, edgeN_mean, edgeN_std, edge_feat, E);
  fc_small_kernel<<<(E * 128 + 255) / 256, 256, 0, stream>>>(
      edge_feat, (const float*)d_in[EE + 0], (const float*)d_in[EE + 3], h1, E, 7);
  mlp2_wmma_kernel<false, true><<<edgeBlocks, 256, 0, stream>>>(
      h1, encE_W2t, (const float*)d_in[EE + 4], encE_W3t, (const float*)d_in[EE + 5],
      (const float*)d_in[EE + 7], (const float*)d_in[EE + 6], edge_lat, E);

  // ---- message passing ----
  for (int i = 0; i < NBLK; ++i) {
    const int BE = P + i * 16, BN = P + i * 16 + 8;
    zero_kernel<<<(N * 128 + 255) / 256, 256, 0, stream>>>(agg, N * 128);
    edge_block_kernel<<<edgeBlocks, 256, 0, stream>>>(
        edge_lat, node_lat, snd, rcv,
        eW1t[i], (const float*)d_in[BE + 3],
        eW2t[i], (const float*)d_in[BE + 4],
        eW3t[i], (const float*)d_in[BE + 5],
        (const float*)d_in[BE + 7], (const float*)d_in[BE + 6],
        edge_lat, agg, E);
    node_block_kernel<<<nodeBlocks, 256, 0, stream>>>(
        node_lat, agg,
        nW1t[i], (const float*)d_in[BN + 3],
        nW2t[i], (const float*)d_in[BN + 4],
        nW3t[i], (const float*)d_in[BN + 5],
        (const float*)d_in[BN + 7], (const float*)d_in[BN + 6],
        node_lat, N);
  }

  // ---- decode + integrate ----
  mlp2_wmma_kernel<true, false><<<nodeBlocks, 256, 0, stream>>>(
      node_lat, dec_W1t, decb0, dec_W2t, decb1, nullptr, nullptr, h1, N);
  decode_integrate_kernel<<<(N + 255) / 256, 256, 0, stream>>>(
      h1, decW2, decb2, outN_mean, outN_std, wp, pwp, ntype, (float*)d_out, N);
}
